// EncoderBlock_70978629533962
// MI455X (gfx1250) — matrix operand order repaired and checked by reading, not compiled, not run on hardware
//
#include <hip/hip_runtime.h>

#define HIDDEN   1024
#define HEADS    16
#define HEAD_DIM 64
#define MLP_DIM  4096
#define BATCH    8
#define SEQ      1024
#define TOKENS   (BATCH*SEQ)

typedef __attribute__((ext_vector_type(16))) _Float16     v16h;
typedef __attribute__((ext_vector_type(8)))  float        v8f;
typedef __attribute__((ext_vector_type(4)))  unsigned int v4u;

union Frag {
  v16h h;
  v4u  q[2];
  unsigned int d[8];
};

__device__ __forceinline__ v8f wmma32(const v16h& a, const v16h& b, const v8f& c) {
  // D = A(16x32 f16) * B(32x16 f16) + C(16x16 f32)
  return __builtin_amdgcn_wmma_f32_16x16x32_f16(false, a, false, b, (short)0, c,
                                                false, false);
}

// ---------------------------------------------------------------------------
// CDNA5 async global->LDS copy (ASYNCcnt-tracked). Falls back to VGPR staging
// if the builtin is not available in this toolchain.
// ---------------------------------------------------------------------------
#if defined(__HIP_DEVICE_COMPILE__) && __has_builtin(__builtin_amdgcn_global_load_async_to_lds_b128)
#define GFX1250_ASYNC_LDS 1
typedef int vi4 __attribute__((vector_size(16)));
typedef __attribute__((address_space(1))) vi4* async_gptr;
typedef __attribute__((address_space(3))) vi4* async_lptr;
#endif

__device__ __forceinline__ void copy16_async(const _Float16* g, _Float16* l) {
#ifdef GFX1250_ASYNC_LDS
  __builtin_amdgcn_global_load_async_to_lds_b128((async_gptr)g, (async_lptr)l, 0, 0);
#else
  *(v4u*)l = *(const v4u*)g;
#endif
}
__device__ __forceinline__ void wait_async_le4() {
#ifdef GFX1250_ASYNC_LDS
  asm volatile("s_wait_asynccnt 4" ::: "memory");
#endif
}
__device__ __forceinline__ void wait_async_0() {
#ifdef GFX1250_ASYNC_LDS
  asm volatile("s_wait_asynccnt 0" ::: "memory");
#endif
}

// ---------------------------------------------------------------------------
// Weight transpose + fp32 -> fp16 convert:  Wt[n][k] = W[k][n]
// ---------------------------------------------------------------------------
__global__ void __launch_bounds__(256) transpose_cvt(const float* __restrict__ W,
                                                     _Float16* __restrict__ Wt,
                                                     int K, int N) {
  __shared__ float tile[32][33];
  const int n0 = blockIdx.x * 32, k0 = blockIdx.y * 32;
  for (int r = threadIdx.y; r < 32; r += 8)
    tile[r][threadIdx.x] = W[(size_t)(k0 + r) * N + n0 + threadIdx.x];
  __syncthreads();
  for (int r = threadIdx.y; r < 32; r += 8)
    Wt[(size_t)(n0 + r) * K + k0 + threadIdx.x] = (_Float16)tile[threadIdx.x][r];
}

// ---------------------------------------------------------------------------
// Row LayerNorm (1024 cols) + fp16 convert
// ---------------------------------------------------------------------------
__global__ void __launch_bounds__(256) layernorm_cvt(const float* __restrict__ X,
                                                     const float* __restrict__ gamma,
                                                     const float* __restrict__ beta,
                                                     _Float16* __restrict__ H) {
  __shared__ float red[256];
  const int row = blockIdx.x, t = threadIdx.x;
  const float* xr = X + (size_t)row * HIDDEN;
  float v[4];
  float s = 0.f;
#pragma unroll
  for (int k = 0; k < 4; ++k) { v[k] = xr[t + 256 * k]; s += v[k]; }
  red[t] = s; __syncthreads();
  for (int o = 128; o > 0; o >>= 1) { if (t < o) red[t] += red[t + o]; __syncthreads(); }
  const float mean = red[0] * (1.f / HIDDEN);
  __syncthreads();
  float s2 = 0.f;
#pragma unroll
  for (int k = 0; k < 4; ++k) { float d = v[k] - mean; s2 += d * d; }
  red[t] = s2; __syncthreads();
  for (int o = 128; o > 0; o >>= 1) { if (t < o) red[t] += red[t + o]; __syncthreads(); }
  const float rstd = rsqrtf(red[0] * (1.f / HIDDEN) + 1e-6f);
#pragma unroll
  for (int k = 0; k < 4; ++k) {
    const int c = t + 256 * k;
    H[(size_t)row * HIDDEN + c] = (_Float16)((v[k] - mean) * rstd * gamma[c] + beta[c]);
  }
}

// ---------------------------------------------------------------------------
// C[M,N](f32) = A[M,K](f16,row-major) x Bt[N,K](f16, pre-transposed weight)
//   + bias, optional exact GELU, optional f32 residual; writes f32 and/or f16.
// 256 threads (8 waves), 128x128 tile, K-step 32, double-buffered async-LDS.
// ---------------------------------------------------------------------------
__global__ void __launch_bounds__(256) gemm_wmma(const _Float16* __restrict__ A,
                                                 const _Float16* __restrict__ Bt,
                                                 const float* __restrict__ bias,
                                                 const float* __restrict__ resid,
                                                 float* __restrict__ outF,
                                                 _Float16* __restrict__ outH,
                                                 int M, int N, int K, int gelu) {
  __shared__ __align__(16) _Float16 As[2][128 * 40];  // 32 K-halves + pad 8
  __shared__ __align__(16) _Float16 Bs[2][128 * 40];
  const int t = threadIdx.x, lane = t & 31, wid = t >> 5;
  const int wm = wid & 3, wn = wid >> 2;            // 4 x 2 wave grid
  const int mBase = blockIdx.y * 128, nBase = blockIdx.x * 128;
  const int lrow = t >> 1, lseg = (t & 1) * 16;     // 16 halves per (row,seg)
  const _Float16* Ag = A  + (size_t)(mBase + lrow) * K + lseg;
  const _Float16* Bg = Bt + (size_t)(nBase + lrow) * K + lseg;
  const int ldso = lrow * 40 + lseg;
  const int r = lane & 15, hi = lane >> 4, g = hi * 8;

  v8f acc[2][4];
#pragma unroll
  for (int mt = 0; mt < 2; ++mt)
#pragma unroll
    for (int nt = 0; nt < 4; ++nt)
      acc[mt][nt] = (v8f){0.f, 0.f, 0.f, 0.f, 0.f, 0.f, 0.f, 0.f};

  // prologue: stage kt = 0 into buffer 0
  copy16_async(Ag,     &As[0][ldso]);
  copy16_async(Ag + 8, &As[0][ldso + 8]);
  copy16_async(Bg,     &Bs[0][ldso]);
  copy16_async(Bg + 8, &Bs[0][ldso + 8]);

  int pb = 0;
  for (int kt = 0; kt < K; kt += 32, pb ^= 1) {
    const int ktn = kt + 32;
    if (ktn < K) {                       // stage next tile into other buffer
      copy16_async(Ag + ktn,     &As[pb ^ 1][ldso]);
      copy16_async(Ag + ktn + 8, &As[pb ^ 1][ldso + 8]);
      copy16_async(Bg + ktn,     &Bs[pb ^ 1][ldso]);
      copy16_async(Bg + ktn + 8, &Bs[pb ^ 1][ldso + 8]);
      __builtin_prefetch(Ag + ktn + 32, 0, 3);   // warm L2 for the tile after
      __builtin_prefetch(Bg + ktn + 32, 0, 3);
      wait_async_le4();                  // in-order: current buffer has landed
    } else {
      wait_async_0();
    }
    __syncthreads();                     // all waves' copies of buf pb visible

    Frag af[2], bf[4];
#pragma unroll
    for (int mt = 0; mt < 2; ++mt) {     // A frag: lane = M row, K contiguous
      const _Float16* ap = &As[pb][(wm * 32 + mt * 16 + r) * 40];
      af[mt].q[0] = *(const v4u*)(ap + hi * 16);
      af[mt].q[1] = *(const v4u*)(ap + hi * 16 + 8);
    }
#pragma unroll
    for (int nt = 0; nt < 4; ++nt) {     // B frag: lane = N col, K contiguous
      const _Float16* bp = &Bs[pb][(wn * 64 + nt * 16 + r) * 40 + hi * 16];
      bf[nt].q[0] = *(const v4u*)(bp);
      bf[nt].q[1] = *(const v4u*)(bp + 8);
    }
#pragma unroll
    for (int mt = 0; mt < 2; ++mt)
#pragma unroll
      for (int nt = 0; nt < 4; ++nt)
        acc[mt][nt] = wmma32(af[mt].h, bf[nt].h, acc[mt][nt]);
    __syncthreads();                     // done reading buf pb; may be rewritten
  }

#pragma unroll
  for (int mt = 0; mt < 2; ++mt)
#pragma unroll
    for (int nt = 0; nt < 4; ++nt) {
      const int col = nBase + wn * 64 + nt * 16 + r;
      const float bc = bias ? bias[col] : 0.f;
#pragma unroll
      for (int i = 0; i < 8; ++i) {
        const int row = mBase + wm * 32 + mt * 16 + hi * 8 + i;
        float v = acc[mt][nt][i] + bc;
        if (gelu) v = 0.5f * v * (1.f + erff(v * 0.70710678118f));
        const size_t idx = (size_t)row * N + col;
        if (resid) v += resid[idx];
        if (outF) outF[idx] = v;
        if (outH) outH[idx] = (_Float16)v;
      }
    }
}

// ---------------------------------------------------------------------------
// Flash attention: one wave owns a 16-row q tile, streams keys 32 at a time.
// scores = Q Kt / 8 (2 WMMAs per 16-key tile), online softmax, ctx += P V
// (4 WMMAs).  P re-laid C->A via per-wave LDS scratch.
// ---------------------------------------------------------------------------
__global__ void __launch_bounds__(256) attention_flash(const _Float16* __restrict__ Q,
                                                       const _Float16* __restrict__ Km,
                                                       const _Float16* __restrict__ V,
                                                       _Float16* __restrict__ O) {
  __shared__ __align__(16) _Float16 Ps[8][16 * 40];   // per-wave P scratch
  const int lane = threadIdx.x & 31, wid = threadIdx.x >> 5;
  const int blk = blockIdx.x;
  const int qb = blk & 7;
  const int h  = (blk >> 3) & 15;
  const int b  = blk >> 7;
  const int qbase = qb * 128 + wid * 16;
  const int r = lane & 15, hi = lane >> 4, g = hi * 8;

  // Q A-fragments (d 0..31 and 32..63) straight from global (table order)
  const _Float16* qrow = Q + (size_t)(b * SEQ + qbase + r) * HIDDEN + h * HEAD_DIM;
  Frag aq0, aq1;
  aq0.q[0] = *(const v4u*)(qrow + g);
  aq0.q[1] = *(const v4u*)(qrow + 16 + g);
  aq1.q[0] = *(const v4u*)(qrow + 32 + g);
  aq1.q[1] = *(const v4u*)(qrow + 48 + g);

  float m[8], l[8];
  v8f cacc[4];
#pragma unroll
  for (int i = 0; i < 8; ++i) { m[i] = -1e30f; l[i] = 0.f; }
#pragma unroll
  for (int j = 0; j < 4; ++j)
    cacc[j] = (v8f){0.f, 0.f, 0.f, 0.f, 0.f, 0.f, 0.f, 0.f};

  _Float16* psw = &Ps[wid][0];
  const v8f z0 = (v8f){0.f, 0.f, 0.f, 0.f, 0.f, 0.f, 0.f, 0.f};

  for (int kb = 0; kb < SEQ; kb += 32) {
    v8f s[2];
#pragma unroll
    for (int kt = 0; kt < 2; ++kt) {   // two 16-key N tiles
      const _Float16* krow =
          Km + (size_t)(b * SEQ + kb + kt * 16 + r) * HIDDEN + h * HEAD_DIM + hi * 8;
      Frag blo, bhi;                   // B frags: K-dim = d, table order
      blo.q[0] = *(const v4u*)(krow);
      blo.q[1] = *(const v4u*)(krow + 16);
      bhi.q[0] = *(const v4u*)(krow + 32);
      bhi.q[1] = *(const v4u*)(krow + 48);
      s[kt] = wmma32(aq1.h, bhi.h, wmma32(aq0.h, blo.h, z0));
    }

    float pl[8], ph[8];
#pragma unroll
    for (int i = 0; i < 8; ++i) {      // online softmax per q-row
      const float a = s[0][i] * 0.125f;
      const float c = s[1][i] * 0.125f;
      float mx = fmaxf(a, c);
#pragma unroll
      for (int msk = 1; msk < 16; msk <<= 1) mx = fmaxf(mx, __shfl_xor(mx, msk, 32));
      const float mnew  = fmaxf(m[i], mx);
      const float scale = __expf(m[i] - mnew);
      const float ea = __expf(a - mnew);
      const float ec = __expf(c - mnew);
      float rs = ea + ec;
#pragma unroll
      for (int msk = 1; msk < 16; msk <<= 1) rs += __shfl_xor(rs, msk, 32);
      l[i] = l[i] * scale + rs;
      m[i] = mnew;
#pragma unroll
      for (int j = 0; j < 4; ++j) cacc[j][i] *= scale;
      pl[i] = ea; ph[i] = ec;
    }

    // P (C layout) -> LDS -> A layout fragment
#pragma unroll
    for (int i = 0; i < 8; ++i) {
      psw[(hi * 8 + i) * 40 + r]      = (_Float16)pl[i];
      psw[(hi * 8 + i) * 40 + 16 + r] = (_Float16)ph[i];
    }
    asm volatile("s_wait_dscnt 0" ::: "memory");   // wave-local DS RAW fence
    Frag pf;
    const _Float16* pr = psw + r * 40;
    pf.q[0] = *(const v4u*)(pr + hi * 16);
    pf.q[1] = *(const v4u*)(pr + hi * 16 + 8);

    // V B-fragments gathered transposed from global (K-dim = key)
#pragma unroll
    for (int j = 0; j < 4; ++j) {
      Frag vf;
      const int d = j * 16 + r;
#pragma unroll
      for (int i = 0; i < 8; ++i) {
        const _Float16* vp =
            V + (size_t)(b * SEQ + kb + hi * 16 + 2 * i) * HIDDEN + h * HEAD_DIM + d;
        const unsigned lo16 = *(const unsigned short*)vp;
        const unsigned hi16 = *(const unsigned short*)(vp + HIDDEN);
        vf.d[i] = lo16 | (hi16 << 16);
      }
      cacc[j] = wmma32(pf.h, vf.h, cacc[j]);
    }
  }

#pragma unroll
  for (int j = 0; j < 4; ++j)
#pragma unroll
    for (int i = 0; i < 8; ++i) {
      const float v = cacc[j][i] / l[i];
      O[(size_t)(b * SEQ + qbase + hi * 8 + i) * HIDDEN + h * HEAD_DIM + j * 16 + r] =
          (_Float16)v;
    }
}

// ---------------------------------------------------------------------------
extern "C" void kernel_launch(void* const* d_in, const int* in_sizes, int n_in,
                              void* d_out, int out_size, void* d_ws, size_t ws_size,
                              hipStream_t stream) {
  (void)in_sizes; (void)n_in; (void)out_size; (void)ws_size;
  const float* x    = (const float*)d_in[0];
  const float* ln1g = (const float*)d_in[1];
  const float* ln1b = (const float*)d_in[2];
  const float* Wq   = (const float*)d_in[3];
  const float* bq   = (const float*)d_in[4];
  const float* Wk   = (const float*)d_in[5];
  const float* bk   = (const float*)d_in[6];
  const float* Wv   = (const float*)d_in[7];
  const float* bv   = (const float*)d_in[8];
  const float* Wo   = (const float*)d_in[9];
  const float* bo   = (const float*)d_in[10];
  const float* ln2g = (const float*)d_in[11];
  const float* ln2b = (const float*)d_in[12];
  const float* W1   = (const float*)d_in[13];
  const float* b1   = (const float*)d_in[14];
  const float* W2   = (const float*)d_in[15];
  const float* b2   = (const float*)d_in[16];
  float* out = (float*)d_out;

  char* ws = (char*)d_ws;
  size_t off = 0;
  auto take = [&](size_t bytes) -> char* {
    char* p = ws + off;
    off += (bytes + 255) & ~(size_t)255;
    return p;
  };
  const size_t TD2 = (size_t)TOKENS * HIDDEN * sizeof(_Float16);
  _Float16* h16 = (_Float16*)take(TD2);
  _Float16* q16 = (_Float16*)take(TD2);
  _Float16* k16 = (_Float16*)take(TD2);
  _Float16* v16 = (_Float16*)take(TD2);
  _Float16* c16 = (_Float16*)take(TD2);
  _Float16* h2  = (_Float16*)take(TD2);
  _Float16* a16 = (_Float16*)take((size_t)TOKENS * MLP_DIM * sizeof(_Float16));
  float*    x1  = (float*)take((size_t)TOKENS * HIDDEN * sizeof(float));
  _Float16* wqT = (_Float16*)take((size_t)HIDDEN * HIDDEN * sizeof(_Float16));
  _Float16* wkT = (_Float16*)take((size_t)HIDDEN * HIDDEN * sizeof(_Float16));
  _Float16* wvT = (_Float16*)take((size_t)HIDDEN * HIDDEN * sizeof(_Float16));
  _Float16* woT = (_Float16*)take((size_t)HIDDEN * HIDDEN * sizeof(_Float16));
  _Float16* w1T = (_Float16*)take((size_t)HIDDEN * MLP_DIM * sizeof(_Float16));
  _Float16* w2T = (_Float16*)take((size_t)HIDDEN * MLP_DIM * sizeof(_Float16));

  const dim3 tb(32, 8);
  transpose_cvt<<<dim3(HIDDEN / 32, HIDDEN / 32), tb, 0, stream>>>(Wq, wqT, HIDDEN, HIDDEN);
  transpose_cvt<<<dim3(HIDDEN / 32, HIDDEN / 32), tb, 0, stream>>>(Wk, wkT, HIDDEN, HIDDEN);
  transpose_cvt<<<dim3(HIDDEN / 32, HIDDEN / 32), tb, 0, stream>>>(Wv, wvT, HIDDEN, HIDDEN);
  transpose_cvt<<<dim3(HIDDEN / 32, HIDDEN / 32), tb, 0, stream>>>(Wo, woT, HIDDEN, HIDDEN);
  transpose_cvt<<<dim3(MLP_DIM / 32, HIDDEN / 32), tb, 0, stream>>>(W1, w1T, HIDDEN, MLP_DIM);
  transpose_cvt<<<dim3(HIDDEN / 32, MLP_DIM / 32), tb, 0, stream>>>(W2, w2T, MLP_DIM, HIDDEN);

  // LN1 -> h
  layernorm_cvt<<<TOKENS, 256, 0, stream>>>(x, ln1g, ln1b, h16);

  // q/k/v = h @ W{q,k,v} + b
  const dim3 gD(HIDDEN / 128, TOKENS / 128);
  gemm_wmma<<<gD, 256, 0, stream>>>(h16, wqT, bq, nullptr, nullptr, q16,
                                    TOKENS, HIDDEN, HIDDEN, 0);
  gemm_wmma<<<gD, 256, 0, stream>>>(h16, wkT, bk, nullptr, nullptr, k16,
                                    TOKENS, HIDDEN, HIDDEN, 0);
  gemm_wmma<<<gD, 256, 0, stream>>>(h16, wvT, bv, nullptr, nullptr, v16,
                                    TOKENS, HIDDEN, HIDDEN, 0);

  // flash attention -> ctx
  attention_flash<<<BATCH * HEADS * (SEQ / 128), 256, 0, stream>>>(q16, k16, v16, c16);

  // x1 = x + ctx @ Wo + bo
  gemm_wmma<<<gD, 256, 0, stream>>>(c16, woT, bo, x, x1, nullptr,
                                    TOKENS, HIDDEN, HIDDEN, 0);

  // LN2 -> h2
  layernorm_cvt<<<TOKENS, 256, 0, stream>>>(x1, ln2g, ln2b, h2);

  // act = gelu(h2 @ W1 + b1)
  gemm_wmma<<<dim3(MLP_DIM / 128, TOKENS / 128), 256, 0, stream>>>(
      h2, w1T, b1, nullptr, nullptr, a16, TOKENS, MLP_DIM, HIDDEN, 1);

  // out = x1 + act @ W2 + b2
  gemm_wmma<<<gD, 256, 0, stream>>>(a16, w2T, b2, x1, out, nullptr,
                                    TOKENS, HIDDEN, MLP_DIM, 0);
}
